// STACAME_light_77644418777393
// MI455X (gfx1250) — matrix operand order repaired and checked by reading, not compiled, not run on hardware
//
#include <hip/hip_runtime.h>
#include <hip/hip_bf16.h>
#include <stdint.h>

#define IN_DIM 512
#define OUT_DIM 64
#define NEG_SLOPE 0.2f

typedef __attribute__((ext_vector_type(16))) __bf16 v16bf;
typedef __attribute__((ext_vector_type(8)))  __bf16 v8bf;
typedef __attribute__((ext_vector_type(8)))  float  v8f;

static __device__ __forceinline__ __bf16 f2bf(float f) {
  unsigned u = __float_as_uint(f);
  unsigned r = (u + 0x7FFFu + ((u >> 16) & 1u)) >> 16;   // round-to-nearest-even
  unsigned short s = (unsigned short)r;
  return __builtin_bit_cast(__bf16, s);
}

// ---------------------------------------------------------------------------
// 0) zero the aggregation buffer (h1 region of d_out), segment-max keys, denom
// ---------------------------------------------------------------------------
__global__ void k_init(float* __restrict__ out_h1, unsigned* __restrict__ emaxKey,
                       float* __restrict__ denom, int n) {
  int i = blockIdx.x * blockDim.x + threadIdx.x;
  if (i < n * OUT_DIM) out_h1[i] = 0.0f;
  if (i < n) { emaxKey[i] = 0u; denom[i] = 0.0f; }
}

// ---------------------------------------------------------------------------
// 1) convert W (fp32 [512][64]) into WMMA B-fragment layouts (bf16):
//    WbFrag: B = W       (K=512 -> 16 k-chunks, 4 col tiles)   for xp = X @ W
//    WtFrag: B = W^T     (K=64  -> 2 k-chunks, 8 groups x 4 col tiles) for h4
//    B 32x16 bf16 layout per ISA: lanes 0-15 hold K=0..7,16..23 (col = lane&15),
//    lanes 16-31 hold K=8..15,24..31; 16 bf16 per lane, contiguous per fragment.
// ---------------------------------------------------------------------------
__global__ void k_convert_w(const float* __restrict__ W,
                            __bf16* __restrict__ WbFrag,
                            __bf16* __restrict__ WtFrag) {
  int tid = blockIdx.x * blockDim.x + threadIdx.x;
  if (tid < 32768) {                       // WbFrag: idx = ((kc*4+t)*32+lane)*16+j
    int j = tid & 15, lane = (tid >> 4) & 31, f = tid >> 9;
    int kc = f >> 2, t = f & 3;
    int half = lane >> 4, l = lane & 15;
    int K = kc * 32 + (half ? 8 : 0) + (j < 8 ? j : 8 + j);
    int col = t * 16 + l;
    WbFrag[tid] = f2bf(W[(size_t)K * OUT_DIM + col]);
  } else if (tid < 65536) {                // WtFrag: idx = (((g*2+kc)*4+t)*32+lane)*16+j
    int idx = tid - 32768;
    int j = idx & 15, lane = (idx >> 4) & 31, f = idx >> 9;
    int g = f >> 3, kc = (f >> 2) & 1, t = f & 3;
    int half = lane >> 4, l = lane & 15;
    int K = kc * 32 + (half ? 8 : 0) + (j < 8 ? j : 8 + j);       // 0..63  (channel dim of h1)
    int ncol = g * 64 + t * 16 + l;              // 0..511 (output dim of h4)
    WtFrag[idx] = f2bf(W[(size_t)ncol * OUT_DIM + K]);  // W^T[K][ncol] = W[ncol][K]
  }
}

// ---------------------------------------------------------------------------
// 2) xp = X @ W  (bf16 WMMA, fp32 accumulate). One wave = 16 rows x 64 cols.
//    Fused: e_src[row] = xp[row].att_src, e_dst[row] = xp[row].att_dst
//    via 16-lane xor-shuffle reduction.
// ---------------------------------------------------------------------------
__global__ void __launch_bounds__(256) k_gemm1(
    const float* __restrict__ X, const __bf16* __restrict__ WbFrag,
    const float* __restrict__ Asrc, const float* __restrict__ Adst,
    float* __restrict__ xp, float* __restrict__ esrc, float* __restrict__ edst,
    int n_tiles) {
  int wv = (blockIdx.x * blockDim.x + threadIdx.x) >> 5;
  if (wv >= n_tiles) return;
  int lane = threadIdx.x & 31;
  int half = lane >> 4, l = lane & 15;
  int m0 = wv * 16;

  v8f acc[4] = {};
  const float* xr = X + (size_t)(m0 + l) * IN_DIM;
  const int koff = half ? 8 : 0;

  for (int kc = 0; kc < IN_DIM / 32; ++kc) {
    int k0 = kc * 32;
    // A 16x32 bf16 fragment: elems 0..7 -> K=k0+koff+0..7, 8..15 -> K=k0+16+koff+0..7
    float4 f0 = *(const float4*)(xr + k0 + koff);
    float4 f1 = *(const float4*)(xr + k0 + koff + 4);
    float4 f2 = *(const float4*)(xr + k0 + 16 + koff);
    float4 f3 = *(const float4*)(xr + k0 + 20 + koff);
    v16bf a;
    a[0]  = f2bf(f0.x); a[1]  = f2bf(f0.y); a[2]  = f2bf(f0.z); a[3]  = f2bf(f0.w);
    a[4]  = f2bf(f1.x); a[5]  = f2bf(f1.y); a[6]  = f2bf(f1.z); a[7]  = f2bf(f1.w);
    a[8]  = f2bf(f2.x); a[9]  = f2bf(f2.y); a[10] = f2bf(f2.z); a[11] = f2bf(f2.w);
    a[12] = f2bf(f3.x); a[13] = f2bf(f3.y); a[14] = f2bf(f3.z); a[15] = f2bf(f3.w);
#pragma unroll
    for (int t = 0; t < 4; ++t) {
      const v16bf b = *(const v16bf*)(WbFrag + (size_t)((kc * 4 + t) * 32 + lane) * 16);
      acc[t] = __builtin_amdgcn_wmma_f32_16x16x32_bf16(
          false, a, false, b, (short)0, acc[t], false, false);
    }
  }

  // store xp: C/D layout: VGPR r -> row m0 + (half?8:0) + r, col = t*16 + l
  int rbase = m0 + (half ? 8 : 0);
#pragma unroll
  for (int t = 0; t < 4; ++t)
#pragma unroll
    for (int r = 0; r < 8; ++r)
      xp[(size_t)(rbase + r) * OUT_DIM + t * 16 + l] = acc[t][r];

  // fused attention scores
  float as_[4], ad_[4];
#pragma unroll
  for (int t = 0; t < 4; ++t) { as_[t] = Asrc[t * 16 + l]; ad_[t] = Adst[t * 16 + l]; }
  float ps[8], pd[8];
#pragma unroll
  for (int r = 0; r < 8; ++r) {
    float s = 0.f, d = 0.f;
#pragma unroll
    for (int t = 0; t < 4; ++t) { s += acc[t][r] * as_[t]; d += acc[t][r] * ad_[t]; }
    ps[r] = s; pd[r] = d;
  }
#pragma unroll
  for (int m = 1; m <= 8; m <<= 1)
#pragma unroll
    for (int r = 0; r < 8; ++r) {
      ps[r] += __shfl_xor(ps[r], m, 32);
      pd[r] += __shfl_xor(pd[r], m, 32);
    }
  if (l == 0)
#pragma unroll
    for (int r = 0; r < 8; ++r) { esrc[rbase + r] = ps[r]; edst[rbase + r] = pd[r]; }
}

// ---------------------------------------------------------------------------
// order-preserving float<->uint key for atomic segment-max
// ---------------------------------------------------------------------------
static __device__ __forceinline__ unsigned f2key(float f) {
  unsigned u = __float_as_uint(f);
  return u ^ ((unsigned)((int)u >> 31) | 0x80000000u);
}
static __device__ __forceinline__ float key2f(unsigned k) {
  unsigned u = (k & 0x80000000u) ? (k ^ 0x80000000u) : ~k;
  return __uint_as_float(u);
}

// 3) per-edge logit + segment max over dst
__global__ void k_edge_logit(const int* __restrict__ src, const int* __restrict__ dst,
                             const float* __restrict__ esrc, const float* __restrict__ edst,
                             float* __restrict__ logit, unsigned* __restrict__ emaxKey,
                             int ne) {
  int e = blockIdx.x * blockDim.x + threadIdx.x;
  if (e >= ne) return;
  float x = esrc[src[e]] + edst[dst[e]];
  float v = x > 0.f ? x : NEG_SLOPE * x;
  logit[e] = v;
  atomicMax(&emaxKey[dst[e]], f2key(v));
}

// 4) w = exp(logit - max[dst]); denom = segment_sum(w)  (in-place over logit buf)
__global__ void k_edge_exp(const int* __restrict__ dst, float* __restrict__ wbuf,
                           const unsigned* __restrict__ emaxKey,
                           float* __restrict__ denom, int ne) {
  int e = blockIdx.x * blockDim.x + threadIdx.x;
  if (e >= ne) return;
  int d = dst[e];
  float w = __expf(wbuf[e] - key2f(emaxKey[d]));
  wbuf[e] = w;
  atomicAdd(&denom[d], w);
}

// 5) out[dst] += alpha * xp[src] ; 16 lanes/edge, float4 per lane
__global__ void k_edge_scatter(const int* __restrict__ src, const int* __restrict__ dst,
                               const float* __restrict__ wbuf, const float* __restrict__ denom,
                               const float* __restrict__ xp, float* __restrict__ out, int ne) {
  int tid = blockIdx.x * blockDim.x + threadIdx.x;
  int e = tid >> 4;
  if (e >= ne) return;
  int c = (tid & 15) * 4;
  int s = src[e], d = dst[e];
  float alpha = wbuf[e] / fmaxf(denom[d], 1e-16f);
  float4 v = *(const float4*)(xp + (size_t)s * OUT_DIM + c);
  float* o = out + (size_t)d * OUT_DIM + c;
  atomicAdd(o + 0, alpha * v.x);
  atomicAdd(o + 1, alpha * v.y);
  atomicAdd(o + 2, alpha * v.z);
  atomicAdd(o + 3, alpha * v.w);
}

// 6) ELU in place on h1; also emit bf16 copy for GEMM2 A-operand
__global__ void k_elu(float* __restrict__ h1, __bf16* __restrict__ h1b, int n64) {
  int i = blockIdx.x * blockDim.x + threadIdx.x;
  if (i >= n64) return;
  float x = h1[i];
  float y = x > 0.f ? x : (__expf(x) - 1.f);
  h1[i] = y;
  h1b[i] = f2bf(y);
}

// ---------------------------------------------------------------------------
// 7) h4 = h1 @ W^T  (bf16 WMMA). Wave = 16 rows x 64 cols; 8 col-groups cover 512.
// ---------------------------------------------------------------------------
__global__ void __launch_bounds__(256) k_gemm2(
    const __bf16* __restrict__ h1b, const __bf16* __restrict__ WtFrag,
    float* __restrict__ h4, int n_tiles) {
  int wvg = (blockIdx.x * blockDim.x + threadIdx.x) >> 5;
  int rt = wvg >> 3, g = wvg & 7;
  if (rt >= n_tiles) return;
  int lane = threadIdx.x & 31;
  int half = lane >> 4, l = lane & 15;
  int m0 = rt * 16;

  v8f acc[4] = {};
  const __bf16* ar = h1b + (size_t)(m0 + l) * OUT_DIM;
#pragma unroll
  for (int kc = 0; kc < 2; ++kc) {
    int koff = kc * 32 + (half ? 8 : 0);
    v8bf a0 = *(const v8bf*)(ar + koff);
    v8bf a1 = *(const v8bf*)(ar + koff + 16);
    v16bf a;
#pragma unroll
    for (int j = 0; j < 8; ++j) { a[j] = a0[j]; a[8 + j] = a1[j]; }
#pragma unroll
    for (int t = 0; t < 4; ++t) {
      const v16bf b = *(const v16bf*)(WtFrag + (size_t)(((g * 2 + kc) * 4 + t) * 32 + lane) * 16);
      acc[t] = __builtin_amdgcn_wmma_f32_16x16x32_bf16(
          false, a, false, b, (short)0, acc[t], false, false);
    }
  }
  int rbase = m0 + (half ? 8 : 0);
#pragma unroll
  for (int t = 0; t < 4; ++t)
#pragma unroll
    for (int r = 0; r < 8; ++r)
      h4[(size_t)(rbase + r) * IN_DIM + g * 64 + t * 16 + l] = acc[t][r];
}

// ---------------------------------------------------------------------------
extern "C" void kernel_launch(void* const* d_in, const int* in_sizes, int n_in,
                              void* d_out, int out_size, void* d_ws, size_t ws_size,
                              hipStream_t stream) {
  const float* X    = (const float*)d_in[0];
  const int*   EI   = (const int*)d_in[1];
  const float* W    = (const float*)d_in[2];
  const float* Asrc = (const float*)d_in[3];
  const float* Adst = (const float*)d_in[4];

  const int n  = in_sizes[0] / IN_DIM;   // 100000
  const int ne = in_sizes[1] / 2;        // 1600000
  const int* src = EI;
  const int* dst = EI + ne;

  // workspace carve-out (256B aligned)
  char* ws = (char*)d_ws;
  size_t off = 0;
  auto carve = [&](size_t bytes) -> void* {
    void* p = ws + off;
    off = (off + bytes + 255) & ~(size_t)255;
    return p;
  };
  float*    xp     = (float*)   carve((size_t)n * OUT_DIM * 4);
  __bf16*   h1b    = (__bf16*)  carve((size_t)n * OUT_DIM * 2);
  float*    esrc   = (float*)   carve((size_t)n * 4);
  float*    edst   = (float*)   carve((size_t)n * 4);
  unsigned* emk    = (unsigned*)carve((size_t)n * 4);
  float*    denom  = (float*)   carve((size_t)n * 4);
  float*    wbuf   = (float*)   carve((size_t)ne * 4);
  __bf16*   WbFrag = (__bf16*)  carve(65536);
  __bf16*   WtFrag = (__bf16*)  carve(65536);

  float* out_h1 = (float*)d_out;
  float* out_h4 = out_h1 + (size_t)n * OUT_DIM;

  const int RT = n / 16;                 // 6250 row tiles (N % 16 == 0)
  const int n64 = n * OUT_DIM;

  k_init<<<dim3((n64 + 255) / 256), dim3(256), 0, stream>>>(out_h1, emk, denom, n);
  k_convert_w<<<dim3(256), dim3(256), 0, stream>>>(W, WbFrag, WtFrag);
  k_gemm1<<<dim3((RT + 7) / 8), dim3(256), 0, stream>>>(X, WbFrag, Asrc, Adst,
                                                        xp, esrc, edst, RT);
  k_edge_logit<<<dim3((ne + 255) / 256), dim3(256), 0, stream>>>(src, dst, esrc, edst,
                                                                 wbuf, emk, ne);
  k_edge_exp<<<dim3((ne + 255) / 256), dim3(256), 0, stream>>>(dst, wbuf, emk, denom, ne);
  k_edge_scatter<<<dim3((ne * 16 + 255) / 256), dim3(256), 0, stream>>>(src, dst, wbuf,
                                                                        denom, xp, out_h1, ne);
  k_elu<<<dim3((n64 + 255) / 256), dim3(256), 0, stream>>>(out_h1, h1b, n64);
  k_gemm2<<<dim3(RT), dim3(256), 0, stream>>>(h1b, WtFrag, out_h4, RT);
}
